// selfLSTM_2_20684562497777
// MI455X (gfx1250) — matrix operand order repaired and checked by reading, not compiled, not run on hardware
//
#include <hip/hip_runtime.h>

// MI455X (gfx1250) implementation of the LSTM+attention reference.
// Roofline: ~28 GFLOP over ~20MB -> compute/latency bound, dominated by the
// 192 sequential scan steps. Strategy: batched bf16 WMMA GEMMs for all
// parallel work (input projections, Y_Wy), and persistent single-workgroup
// scan kernels that keep h (LDS bf16) / c (registers) resident and use
// v_wmma_f32_16x16x32_bf16 (K=32/instr, 8x fewer serial matrix instructions
// than the f32 16x16x4 path) with f32 accumulation + f32 nonlinearities.
// Embedding rows / Wih are pre-padded to K=128 bf16 so every WMMA operand is
// a branch-free aligned 16B/32B vector load.  XZ (the precomputed input
// projection) is kept in WMMA C/D tile-register layout [tile][lane][8] so the
// producer stores and the LSTM's per-step accumulator loads are contiguous
// b128 operations instead of stride-8KB scalar traffic.

typedef __bf16 bf16_t;
typedef __attribute__((ext_vector_type(16))) __bf16 v16bf;
typedef __attribute__((ext_vector_type(8)))  float  v8f;

struct B32 { uint4 lo, hi; };
struct F8  { float4 lo, hi; };

__device__ __forceinline__ bf16_t f2bf(float f) {
  unsigned u = __builtin_bit_cast(unsigned, f);
  unsigned r = (u + 0x7FFFu + ((u >> 16) & 1u)) >> 16;   // round-nearest-even
  unsigned short s = (unsigned short)r;
  return __builtin_bit_cast(bf16_t, s);
}
__device__ __forceinline__ bf16_t bf_zero() {
  return __builtin_bit_cast(bf16_t, (unsigned short)0);
}

// B-matrix tile (32x16 bf16): lane gives N=lane&15; per lane 16 contiguous
// K-values starting at kc*32 + 16*(lane>>4).  One 32B load.
__device__ __forceinline__ v16bf load_b16(const bf16_t* p) {
  B32 t; t.lo = *(const uint4*)p; t.hi = *(const uint4*)(p + 8);
  return __builtin_bit_cast(v16bf, t);
}
// A-matrix tile (16x32 bf16): lane gives M=lane&15; element j holds
// K = j + 16*halfL  =>  16 contiguous K-values (one 32B load).
__device__ __forceinline__ v16bf load_a16(const bf16_t* row, int kc, int halfL) {
  const bf16_t* p = row + kc * 32 + 16 * halfL;
  B32 t; t.lo = *(const uint4*)p; t.hi = *(const uint4*)(p + 8);
  return __builtin_bit_cast(v16bf, t);
}
__device__ __forceinline__ v8f load_f8(const float* p) {   // 32B, two b128
  F8 t; t.lo = *(const float4*)p; t.hi = *(const float4*)(p + 4);
  return __builtin_bit_cast(v8f, t);
}
__device__ __forceinline__ void store_f8(float* p, v8f v) {
  F8 t = __builtin_bit_cast(F8, v);
  *(float4*)p = t.lo; *(float4*)(p + 4) = t.hi;
}
__device__ __forceinline__ v8f wmma_bf(v16bf a, v16bf b, v8f c) {
  return __builtin_amdgcn_wmma_f32_16x16x32_bf16(false, a, false, b, (short)0, c,
                                                 false, false);
}
__device__ __forceinline__ float sigm(float x) { return 1.0f / (1.0f + __expf(-x)); }

// ---------------------------------------------------------------- weights->bf16
__global__ void cvt_bf16_kernel(const float* __restrict__ src, bf16_t* __restrict__ dst, int n) {
  int i = blockIdx.x * blockDim.x + threadIdx.x;
  if (i < n) dst[i] = f2bf(src[i]);
}

// ------------------------------- gather embedding rows, zero-padded 100 -> 128
__global__ void gather_pad_kernel(const int* __restrict__ xs1, const int* __restrict__ xs2,
                                  const float* __restrict__ embed, bf16_t* __restrict__ Exb) {
  int row = blockIdx.x;                         // 8192 rows (seq0 then seq1)
  int e = threadIdx.x;                          // 128 threads
  int idx = (row < 4096) ? xs1[row] : xs2[row - 4096];
  float v = (e < 100) ? embed[(size_t)idx * 100 + e] : 0.0f;
  Exb[(size_t)row * 128 + e] = f2bf(v);
}

// ------------------------------------- pad Wih (2048x100 f32) -> 2048x128 bf16
__global__ void pad_wih_kernel(const float* __restrict__ WihL, const float* __restrict__ WihR,
                               bf16_t* __restrict__ dstL, bf16_t* __restrict__ dstR) {
  int row = blockIdx.x;                         // 4096 rows (L then R)
  int e = threadIdx.x;                          // 128 threads
  const float* src = (row < 2048) ? WihL : WihR;
  bf16_t* dst = (row < 2048) ? dstL : dstR;
  int r = row & 2047;
  float v = (e < 100) ? src[(size_t)r * 100 + e] : 0.0f;
  dst[(size_t)r * 128 + e] = f2bf(v);
}

// --------------------------------------------------- XZ = embed[x] @ Wih.T + b
// Rows = (t,b) flattened (8192 over both sequences), cols = 2048 gates.
// One 16x16 tile per wave; K=128 padded bf16 operands, 4 WMMA chunks.
// Output kept in tile-register layout: XZ[tile=(rt&255)*128+jt][lane][v].
__global__ void xz_kernel(const bf16_t* __restrict__ Exb,
                          const bf16_t* __restrict__ WihLb, const bf16_t* __restrict__ WihRb,
                          const float* __restrict__ bL, const float* __restrict__ bR,
                          float* __restrict__ XZ1, float* __restrict__ XZ2) {
  int tid = threadIdx.x, lane = tid & 31, wave = tid >> 5;
  int halfL = lane >> 4, lm = lane & 15;
  int id = blockIdx.x * 8 + wave;          // 65536 tiles
  int rt = id >> 7, jt = id & 127;
  int seq = rt >= 256;
  const bf16_t* Wih  = seq ? WihRb : WihLb;
  const float*  bias = seq ? bR    : bL;
  float*        XZ   = seq ? XZ2   : XZ1;

  const bf16_t* arow = Exb + (size_t)(rt * 16 + lm) * 128;     // A row (global row id)
  int jc = jt * 16 + lm;
  const bf16_t* wrow = Wih + (size_t)jc * 128;                 // B row
  float bv = bias[jc];
  v8f acc = {bv, bv, bv, bv, bv, bv, bv, bv};

#pragma unroll
  for (int kc = 0; kc < 4; ++kc)
    acc = wmma_bf(load_a16(arow, kc, halfL),
                  load_b16(wrow + kc * 32 + 16 * halfL), acc);

  size_t tl = (size_t)(rt & 255) * 128 + jt;     // tile-major, raw C/D layout
  store_f8(XZ + tl * 256 + lane * 8, acc);
}

// --------------------------------------------------------- persistent dual LSTM
// One workgroup, 32 waves.  Each wave owns 4 tile-quads (btile, kk-tile) and
// keeps its cell state c in registers across all 128 steps; h is shared via LDS.
__global__ __launch_bounds__(1024) void lstm_kernel(
    const float* __restrict__ XZ1, const float* __restrict__ XZ2,
    const bf16_t* __restrict__ WhhL, const bf16_t* __restrict__ WhhR,
    float* __restrict__ Y, bf16_t* __restrict__ Ybf, bf16_t* __restrict__ Yrbf,
    float* __restrict__ hn) {
  extern __shared__ char smem[];
  bf16_t* h_lds = (bf16_t*)smem;                 // [64][512] bf16 = 64KB
  int tid = threadIdx.x, lane = tid & 31, wave = tid >> 5;
  int halfL = lane >> 4, lm = lane & 15;
  int btile = wave >> 3;
  int kkbase = (wave & 7) * 4;

  for (int i = tid; i < 64 * 512; i += 1024) h_lds[i] = bf_zero();
  float creg[4][8] = {};
  __syncthreads();

  for (int seq = 0; seq < 2; ++seq) {
    const float*  XZ  = seq ? XZ2   : XZ1;
    const bf16_t* Whh = seq ? WhhR  : WhhL;
    bf16_t*       Yb  = seq ? Yrbf  : Ybf;
    for (int t = 0; t < 64; ++t) {
      float hnew[4][8];
      const bf16_t* hrow = &h_lds[(btile * 16 + lm) * 512];
      // xz tiles for this step, tile id = (t*4 + btile)*128 + g*32 + kkt
      const float* xzb = XZ + ((size_t)(t * 4 + btile) * 128) * 256 + lane * 8;
      for (int q = 0; q < 4; ++q) {
        int kkt = kkbase + q;
        v8f acc[4];
#pragma unroll
        for (int g = 0; g < 4; ++g)
          acc[g] = load_f8(xzb + (size_t)(g * 32 + kkt) * 256);
        for (int kc = 0; kc < 16; ++kc) {
          v16bf a = load_a16(hrow, kc, halfL);
#pragma unroll
          for (int g = 0; g < 4; ++g) {
            const bf16_t* wrow = Whh + (size_t)(g * 512 + kkt * 16 + lm) * 512
                                 + kc * 32 + 16 * halfL;
            acc[g] = wmma_bf(a, load_b16(wrow), acc[g]);
          }
        }
#pragma unroll
        for (int v = 0; v < 8; ++v) {      // torch gate order (i,f,g,o)
          float zi = acc[0][v], zf = acc[1][v], zg = acc[2][v], zo = acc[3][v];
          float c = sigm(zf) * creg[q][v] + sigm(zi) * tanhf(zg);
          creg[q][v] = c;
          hnew[q][v] = sigm(zo) * tanhf(c);
        }
      }
      __syncthreads();                     // all h reads done before overwrite
      for (int q = 0; q < 4; ++q) {
        int kkt = kkbase + q;
#pragma unroll
        for (int v = 0; v < 8; ++v) {
          int b = btile * 16 + v + 8 * halfL;
          int kk = kkt * 16 + lm;
          float hv = hnew[q][v];
          bf16_t hb = f2bf(hv);
          h_lds[b * 512 + kk] = hb;
          Yb[(size_t)(t * 64 + b) * 512 + kk] = hb;
          if (!seq) Y[(size_t)(t * 64 + b) * 512 + kk] = hv;
          if (seq && t == 63) hn[b * 512 + kk] = hv;
        }
      }
      __syncthreads();
    }
  }
}

// ----------------------------------------------- YWy[b][l][j] = sum_k Wy[j,k]*Y[l,b,k]
__global__ void ywy_kernel(const bf16_t* __restrict__ Wyb, const bf16_t* __restrict__ Ybf,
                           float* __restrict__ YWy) {
  int tid = threadIdx.x, lane = tid & 31, wave = tid >> 5;
  int halfL = lane >> 4, lm = lane & 15;
  int id = blockIdx.x * 8 + wave;          // 8192 tiles: b(64) x jt(32) x lt(4)
  int b = id >> 7, jt = (id >> 2) & 31, lt = id & 3;
  v8f acc = {};
  const bf16_t* arow = Wyb + (size_t)(jt * 16 + lm) * 512;
  for (int kc = 0; kc < 16; ++kc) {
    v16bf a = load_a16(arow, kc, halfL);
    const bf16_t* brow = Ybf + ((size_t)(lt * 16 + lm) * 64 + b) * 512
                         + kc * 32 + 16 * halfL;
    acc = wmma_bf(a, load_b16(brow), acc);
  }
  float* out = YWy + ((size_t)b * 64 + lt * 16 + lm) * 512 + jt * 16 + 8 * halfL;
#pragma unroll
  for (int v = 0; v < 8; ++v) out[v] = acc[v];
}

// -------------------------------------------------------- persistent attention
__global__ __launch_bounds__(1024) void attn_kernel(
    const float* __restrict__ Y, const bf16_t* __restrict__ Yrbf,
    const float* __restrict__ YWy, const bf16_t* __restrict__ Whb,
    const bf16_t* __restrict__ Wrb, const float* __restrict__ w_t,
    float* __restrict__ r_out) {
  extern __shared__ char smem[];
  float*  hWh  = (float*)smem;                      // 64*512 f32 = 128KB
  bf16_t* r_bf = (bf16_t*)(smem + 131072);          // 64*512 bf16 = 64KB
  float*  sbuf = (float*)(smem + 196608);           // 64*64 f32   = 16KB
  float*  wts  = (float*)(smem + 212992);           // 512 f32     = 2KB
  int tid = threadIdx.x, lane = tid & 31, wave = tid >> 5;
  int halfL = lane >> 4, lm = lane & 15;

  for (int i = tid; i < 64 * 512; i += 1024) r_bf[i] = bf_zero();
  if (tid < 512) wts[tid] = w_t[tid];
  float rreg[32] = {};                              // r[b][k], p=i*1024+tid
  __syncthreads();

  for (int t = 0; t < 64; ++t) {
    // hWh = Yr[t] @ Wh.T + r @ Wr.T   (fused dual-WMMA per k-chunk)
    for (int q = 0; q < 4; ++q) {
      int id = wave * 4 + q, bt = id >> 5, nt = id & 31;
      v8f acc = {};
      const bf16_t* yr_row = Yrbf + ((size_t)(t * 64) + bt * 16 + lm) * 512;
      const bf16_t* r_row  = &r_bf[(bt * 16 + lm) * 512];
      for (int kc = 0; kc < 16; ++kc) {
        const bf16_t* w1 = Whb + (size_t)(nt * 16 + lm) * 512 + kc * 32 + 16 * halfL;
        const bf16_t* w2 = Wrb + (size_t)(nt * 16 + lm) * 512 + kc * 32 + 16 * halfL;
        acc = wmma_bf(load_a16(yr_row, kc, halfL), load_b16(w1), acc);
        acc = wmma_bf(load_a16(r_row,  kc, halfL), load_b16(w2), acc);
      }
#pragma unroll
      for (int v = 0; v < 8; ++v)
        hWh[(bt * 16 + v + 8 * halfL) * 512 + nt * 16 + lm] = acc[v];
    }
    __syncthreads();
    // scores s[b][l] = sum_j tanh(hWh[b][j] + YWy[b][l][j]) * w_t[j]
    for (int i = 0; i < 4; ++i) {
      int p = i * 1024 + tid;                       // p = b*64 + l
      const float* yw = YWy + (size_t)p * 512;
      const float* hb = hWh + (p >> 6) * 512;
      float s = 0.0f;
      for (int k = 0; k < 512; ++k) s += tanhf(hb[k] + yw[k]) * wts[k];
      sbuf[p] = s;
    }
    __syncthreads();
    if (tid < 64) {                                 // softmax over l (in place)
      float* row = sbuf + tid * 64;
      float m = row[0];
      for (int l = 1; l < 64; ++l) m = fmaxf(m, row[l]);
      float sum = 0.0f;
      for (int l = 0; l < 64; ++l) { float e = __expf(row[l] - m); row[l] = e; sum += e; }
      float inv = 1.0f / sum;
      for (int l = 0; l < 64; ++l) row[l] *= inv;
    }
    __syncthreads();
    // r_new[b][k] = sum_l alpha[b][l] * Y[l][b][k]
    for (int i = 0; i < 32; ++i) {
      int p = i * 1024 + tid, b = p >> 9, k = p & 511;
      const float* al = sbuf + b * 64;
      float acc = 0.0f;
      for (int l = 0; l < 64; ++l)
        acc += al[l] * Y[((size_t)l * 64 + b) * 512 + k];
      rreg[i] = acc;
    }
    __syncthreads();
    for (int i = 0; i < 32; ++i) {
      int p = i * 1024 + tid;
      r_bf[p] = f2bf(rreg[i]);                      // p == b*512+k
    }
    __syncthreads();
  }
  for (int i = 0; i < 32; ++i) r_out[i * 1024 + tid] = rreg[i];
}

// --------------------------------------------------------------- output head
__global__ void final1_kernel(const float* __restrict__ r, const float* __restrict__ hn,
                              const float* __restrict__ Wp, const float* __restrict__ Wx,
                              float* __restrict__ hout) {
  int idx = blockIdx.x * blockDim.x + threadIdx.x;  // 32768
  int b = idx >> 9, kk = idx & 511;
  const float* rr = r + (size_t)b * 512;
  const float* hh = hn + (size_t)b * 512;
  const float* wp = Wp + (size_t)kk * 512;
  const float* wx = Wx + (size_t)kk * 512;
  float s = 0.0f;
  for (int k = 0; k < 512; ++k) s += rr[k] * wp[k] + hh[k] * wx[k];
  hout[idx] = tanhf(s);
}

__global__ void final2_kernel(const float* __restrict__ hout, const float* __restrict__ Wo,
                              const float* __restrict__ bo, float* __restrict__ out) {
  int tid = threadIdx.x;
  if (tid >= 192) return;
  int b = tid / 3, o = tid % 3;
  const float* hb = hout + (size_t)b * 512;
  const float* wo = Wo + (size_t)o * 512;
  float s = bo[o];
  for (int k = 0; k < 512; ++k) s += hb[k] * wo[k];
  out[tid] = s;
}

extern "C" void kernel_launch(void* const* d_in, const int* in_sizes, int n_in,
                              void* d_out, int out_size, void* d_ws, size_t ws_size,
                              hipStream_t stream) {
  (void)in_sizes; (void)n_in; (void)out_size; (void)ws_size;
  const int*   xs1  = (const int*)d_in[0];
  const int*   xs2  = (const int*)d_in[1];
  const float* embed= (const float*)d_in[2];
  const float* WihL = (const float*)d_in[3];
  const float* WhhL = (const float*)d_in[4];
  const float* bL   = (const float*)d_in[5];
  const float* WihR = (const float*)d_in[6];
  const float* WhhR = (const float*)d_in[7];
  const float* bR   = (const float*)d_in[8];
  const float* Wy   = (const float*)d_in[9];
  const float* Wh   = (const float*)d_in[10];
  const float* Wr   = (const float*)d_in[11];
  const float* wt   = (const float*)d_in[12];
  const float* Wx   = (const float*)d_in[13];
  const float* Wp   = (const float*)d_in[14];
  const float* Wo   = (const float*)d_in[15];
  const float* bo   = (const float*)d_in[16];
  float* out = (float*)d_out;

  char* ws = (char*)d_ws;
  size_t off = 0;
  auto alloc = [&](size_t bytes) -> char* {
    char* p = ws + off;
    off += (bytes + 255) & ~(size_t)255;
    return p;
  };
  float*  XZ1   = (float*)alloc(4096ull * 2048 * 4);
  float*  XZ2   = (float*)alloc(4096ull * 2048 * 4);
  float*  Y     = (float*)alloc(64ull * 64 * 512 * 4);
  bf16_t* Ybf   = (bf16_t*)alloc(64ull * 64 * 512 * 2);
  bf16_t* Yrbf  = (bf16_t*)alloc(64ull * 64 * 512 * 2);
  float*  hn    = (float*)alloc(64ull * 512 * 4);
  float*  YWy   = (float*)alloc(64ull * 64 * 512 * 4);
  float*  rbuf  = (float*)alloc(64ull * 512 * 4);
  float*  hout  = (float*)alloc(64ull * 512 * 4);
  bf16_t* WhhLb = (bf16_t*)alloc(2048ull * 512 * 2);
  bf16_t* WhhRb = (bf16_t*)alloc(2048ull * 512 * 2);
  bf16_t* Wyb   = (bf16_t*)alloc(512ull * 512 * 2);
  bf16_t* Whb   = (bf16_t*)alloc(512ull * 512 * 2);
  bf16_t* Wrb   = (bf16_t*)alloc(512ull * 512 * 2);
  bf16_t* Exb   = (bf16_t*)alloc(8192ull * 128 * 2);
  bf16_t* WihLb = (bf16_t*)alloc(2048ull * 128 * 2);
  bf16_t* WihRb = (bf16_t*)alloc(2048ull * 128 * 2);

  hipFuncSetAttribute(reinterpret_cast<const void*>(lstm_kernel),
                      hipFuncAttributeMaxDynamicSharedMemorySize, 65536);
  hipFuncSetAttribute(reinterpret_cast<const void*>(attn_kernel),
                      hipFuncAttributeMaxDynamicSharedMemorySize, 215040);

  cvt_bf16_kernel<<<(2048 * 512 + 255) / 256, 256, 0, stream>>>(WhhL, WhhLb, 2048 * 512);
  cvt_bf16_kernel<<<(2048 * 512 + 255) / 256, 256, 0, stream>>>(WhhR, WhhRb, 2048 * 512);
  cvt_bf16_kernel<<<(512 * 512 + 255) / 256, 256, 0, stream>>>(Wy, Wyb, 512 * 512);
  cvt_bf16_kernel<<<(512 * 512 + 255) / 256, 256, 0, stream>>>(Wh, Whb, 512 * 512);
  cvt_bf16_kernel<<<(512 * 512 + 255) / 256, 256, 0, stream>>>(Wr, Wrb, 512 * 512);
  gather_pad_kernel<<<8192, 128, 0, stream>>>(xs1, xs2, embed, Exb);
  pad_wih_kernel<<<4096, 128, 0, stream>>>(WihL, WihR, WihLb, WihRb);

  xz_kernel<<<8192, 256, 0, stream>>>(Exb, WihLb, WihRb, bL, bR, XZ1, XZ2);
  lstm_kernel<<<1, 1024, 65536, stream>>>(XZ1, XZ2, WhhLb, WhhRb, Y, Ybf, Yrbf, hn);
  ywy_kernel<<<1024, 256, 0, stream>>>(Wyb, Ybf, YWy);
  attn_kernel<<<1, 1024, 215040, stream>>>(Y, Yrbf, YWy, Whb, Wrb, wt, rbuf);
  final1_kernel<<<128, 256, 0, stream>>>(rbuf, hn, Wp, Wx, hout);
  final2_kernel<<<1, 192, 0, stream>>>(hout, Wo, bo, out);
}
